// DistanceBlock_12558484373711
// MI455X (gfx1250) — matrix operand order repaired and checked by reading, not compiled, not run on hardware
//
#include <hip/hip_runtime.h>
#include <hip/hip_bf16.h>

// ---------------------------------------------------------------------------
// DistanceBlock on gfx1250 (MI455X):
//  - dense GEMMs via v_wmma_f32_16x16x32_f16 (wave32 WMMA, f16 in / f32 acc)
//  - B (weight) tiles staged to LDS with GLOBAL_LOAD_ASYNC_TO_LDS_B128,
//    double-buffered across K-chunks, synced with s_wait_asynccnt
//  - gather attention (per-node neighbour softmax) via VALU + shuffles
// ---------------------------------------------------------------------------

typedef _Float16 f16;
typedef __attribute__((ext_vector_type(16))) _Float16 v16h;
typedef __attribute__((ext_vector_type(8)))  float    v8f;

#define NN   8192
#define KNB  16
#define HH   8
#define DD   128
#define HD   1024      // H*D
#define DIM  128       // FEAT == DIST == HID == 128
#define ATT_SCALE 0.08838834764831845f  // 1/sqrt(128)

// ------------------------------ prep kernels -------------------------------

__global__ void cvt_f32_to_f16(f16* __restrict__ dst, const float* __restrict__ src, int n) {
    int i = blockIdx.x * blockDim.x + threadIdx.x;
    if (i < n) dst[i] = (f16)src[i];
}

// dst (transposed, [O x I] row-major) = (f16) src ([I x O] row-major)
__global__ void transpose_cvt(f16* __restrict__ dst, const float* __restrict__ src, int I, int O) {
    int t = blockIdx.x * blockDim.x + threadIdx.x;
    if (t < I * O) {
        int o = t / I;
        int i = t - o * I;
        dst[t] = (f16)src[(size_t)i * O + o];
    }
}

// ------------------------------ WMMA fragment loaders ----------------------
// A-matrix 16x32 f16 layout (ISA 7.12.2):
//   lanes 0-15 : row M=lane,    K = {0..7, 16..23}
//   lanes 16-31: row M=lane-16, K = {8..15, 24..31}
__device__ __forceinline__ v16h load_a_frag(const f16* __restrict__ base) {
    union { v16h v; float4 f[2]; } u;
    u.f[0] = *(const float4*)(base);        // 8 halves: K = k0+kh*8 .. +7
    u.f[1] = *(const float4*)(base + 16);   // 8 halves: K = k0+16+kh*8 ..
    return u.v;
}

// B-matrix 32x16 f16 layout (sparse-B doc scaled to K=32):
//   lanes 0-15 : col N=lane,    K = {0..7, 16..23}
//   lanes 16-31: col N=lane-16, K = {8..15, 24..31}
// B columns live as contiguous rows of the LDS-staged W^T tile.
__device__ __forceinline__ v16h load_b_frag_lds(const f16* base) {
    union { v16h v; float4 f[2]; } u;
    u.f[0] = *(const float4*)(base);        // K = k0+kh*8 .. +7
    u.f[1] = *(const float4*)(base + 16);   // K = k0+16+kh*8 ..
    return u.v;
}

// ------------------------------ generic GEMM -------------------------------
// C[M x N] = epilogue(A[M x KDIM] @ WT^T + bias (+ resid))
//   A  : f16, row-major, stride KDIM (per-wave rows, straight from global)
//   WT : f16, [N x KDIM] row-major — 64-col tile async-staged into LDS,
//        double buffered over 128-wide K chunks.
// EPI: 0 = store f16; 1 = relu, store f16;
//      2 = + resid(f32), store f32 AND f16; 3 = + resid(f32), store f32 only.
template<int KDIM, int EPI>
__global__ __launch_bounds__(256)
void gemm_wmma(const f16* __restrict__ A, const f16* __restrict__ WT,
               const float* __restrict__ bias, const float* __restrict__ resid,
               float* __restrict__ dstF, f16* __restrict__ dstH,
               int M, int N) {
    constexpr int KCH    = 128;          // K chunk staged per step
    constexpr int CHUNKS = KDIM / KCH;
    static_assert(KDIM % KCH == 0, "KDIM must be a multiple of 128");

    __shared__ __align__(16) f16 Bs[2][64 * KCH];   // 2 x 16 KB double buffer

    const int tid    = threadIdx.x;
    const int lane   = tid & 31;
    const int waveId = tid >> 5;
    const int m0 = blockIdx.x * 128 + waveId * 16;   // 8 waves x 16 rows
    const int n0 = blockIdx.y * 64;                  // 4 column tiles per wave
    const int r  = lane & 15;
    const int kh = lane >> 4;

    const f16* arow = A + (size_t)(m0 + r) * KDIM + kh * 8;   // A-frag base
    __builtin_prefetch(arow, 0, 1);                 // global_prefetch_b8

    // Async-stage one 64 x 128 W^T sub-tile (16 KB) into Bs[buf].
    // 256 threads x 16 B x 4 passes; each wave issues 4 ASYNCcnt ops.
    auto stage = [&](int c, int buf) {
#pragma unroll
        for (int p = 0; p < 4; ++p) {
            const int e   = (p * 256 + tid) * 8;    // half index within tile
            const int row = e >> 7;                 // / KCH
            const int kk  = e & (KCH - 1);
            const unsigned lds = (unsigned)(uintptr_t)&Bs[buf][row * KCH + kk];
            const void* g = (const void*)(WT + (size_t)(n0 + row) * KDIM + c * KCH + kk);
            asm volatile("global_load_async_to_lds_b128 %0, %1, off"
                         :: "v"(lds), "v"(g) : "memory");
        }
    };

    v8f acc[4] = {};

    stage(0, 0);
    for (int c = 0; c < CHUNKS; ++c) {
        const int buf = c & 1;
        if (c + 1 < CHUNKS) {
            stage(c + 1, buf ^ 1);                   // prefetch next chunk
            asm volatile("s_wait_asynccnt 0x4" ::: "memory");  // chunk c landed
        } else {
            asm volatile("s_wait_asynccnt 0x0" ::: "memory");
        }
        __syncthreads();                             // tile visible to all waves

        const f16* abase = arow + c * KCH;
#pragma unroll
        for (int k0 = 0; k0 < KCH; k0 += 32) {
            v16h a = load_a_frag(abase + k0);
#pragma unroll
            for (int t = 0; t < 4; ++t) {
                const f16* bp = &Bs[buf][(16 * t + r) * KCH + kh * 8 + k0];
                v16h b = load_b_frag_lds(bp);
                acc[t] = __builtin_amdgcn_wmma_f32_16x16x32_f16(
                    false, a, false, b, (short)0, acc[t], false, false);
            }
        }
        __syncthreads();                             // done reading buf before overwrite
    }

    // C/D layout: col = lane&15 ; VGPR i holds row = i + 8*(lane>>4)
    const int col  = lane & 15;
    const int rsel = (lane >> 4) * 8;
#pragma unroll
    for (int t = 0; t < 4; ++t) {
        const int cg = n0 + 16 * t + col;
        const float bv = bias[cg];
#pragma unroll
        for (int i = 0; i < 8; ++i) {
            const int rg = m0 + rsel + i;
            float v = acc[t][i] + bv;
            if (EPI == 1) v = v > 0.f ? v : 0.f;
            if (EPI == 2 || EPI == 3) v += resid[(size_t)rg * N + cg];
            if (EPI == 2 || EPI == 3) dstF[(size_t)rg * N + cg] = v;
            if (EPI == 0 || EPI == 1 || EPI == 2) dstH[(size_t)rg * N + cg] = (f16)v;
        }
    }
}

// ------------------------------ attention ----------------------------------
// One wave per (node, head). 32 lanes cover D=128 with 4 elems each.
__global__ __launch_bounds__(256)
void attn_kernel(const f16* __restrict__ qh, const f16* __restrict__ kh,
                 const f16* __restrict__ vh, const int* __restrict__ structure,
                 f16* __restrict__ atth) {
    const int wave = (int)((blockIdx.x * blockDim.x + threadIdx.x) >> 5);
    const int lane = threadIdx.x & 31;
    const int n = wave >> 3;          // node
    const int h = wave & 7;           // head
    if (n >= NN) return;

    const size_t hoff = (size_t)h * DD + lane * 4;
    union V4 { float2 f; f16 h[4]; };

    V4 q; q.f = *(const float2*)(qh + (size_t)n * HD + hoff);
    const float q0 = (float)q.h[0], q1 = (float)q.h[1],
                q2 = (float)q.h[2], q3 = (float)q.h[3];

    int   idx[KNB];
    float w[KNB];

#pragma unroll
    for (int j = 0; j < KNB; ++j) {
        idx[j] = structure[(size_t)n * KNB + j];
        V4 k; k.f = *(const float2*)(kh + (size_t)idx[j] * HD + hoff);
        float s = q0 * (float)k.h[0] + q1 * (float)k.h[1]
                + q2 * (float)k.h[2] + q3 * (float)k.h[3];
#pragma unroll
        for (int m = 16; m >= 1; m >>= 1) s += __shfl_xor(s, m, 32);
        w[j] = s * ATT_SCALE;          // all lanes hold the full logit
    }

    float mx = w[0];
#pragma unroll
    for (int j = 1; j < KNB; ++j) mx = fmaxf(mx, w[j]);
    float den = 0.f;
#pragma unroll
    for (int j = 0; j < KNB; ++j) { w[j] = __expf(w[j] - mx); den += w[j]; }
    const float inv = 1.0f / den;

    float a0 = 0.f, a1 = 0.f, a2 = 0.f, a3 = 0.f;
#pragma unroll
    for (int j = 0; j < KNB; ++j) {
        V4 v; v.f = *(const float2*)(vh + (size_t)idx[j] * HD + hoff);
        const float wj = w[j] * inv;
        a0 += wj * (float)v.h[0]; a1 += wj * (float)v.h[1];
        a2 += wj * (float)v.h[2]; a3 += wj * (float)v.h[3];
    }

    V4 o; o.h[0] = (f16)a0; o.h[1] = (f16)a1; o.h[2] = (f16)a2; o.h[3] = (f16)a3;
    *(float2*)(atth + (size_t)n * HD + hoff) = o.f;
}

// ------------------------------ launch -------------------------------------

extern "C" void kernel_launch(void* const* d_in, const int* in_sizes, int n_in,
                              void* d_out, int out_size, void* d_ws, size_t ws_size,
                              hipStream_t stream) {
    const float* features  = (const float*)d_in[0];
    const float* distf     = (const float*)d_in[1];
    const int*   structure = (const int*)d_in[2];
    const float* Wq = (const float*)d_in[3];  const float* bq = (const float*)d_in[4];
    const float* Wk = (const float*)d_in[5];  const float* bk = (const float*)d_in[6];
    const float* Wv = (const float*)d_in[7];  const float* bv = (const float*)d_in[8];
    const float* Wo = (const float*)d_in[9];  const float* bo = (const float*)d_in[10];
    const float* W1 = (const float*)d_in[11]; const float* b1 = (const float*)d_in[12];
    const float* W2 = (const float*)d_in[13]; const float* b2 = (const float*)d_in[14];
    const float* W3 = (const float*)d_in[15]; const float* b3 = (const float*)d_in[16];
    const float* W4 = (const float*)d_in[17]; const float* b4 = (const float*)d_in[18];

    // ---- workspace carve-up (all offsets 256B aligned) ----
    char* ws = (char*)d_ws;
    size_t off = 0;
    auto take = [&](size_t bytes) { char* p = ws + off; off += (bytes + 255) & ~(size_t)255; return p; };

    f16*   qh    = (f16*)  take((size_t)NN * HD * 2);   // 16 MB
    f16*   khb   = (f16*)  take((size_t)NN * HD * 2);   // 16 MB
    f16*   vhb   = (f16*)  take((size_t)NN * HD * 2);   // 16 MB
    f16*   atth  = (f16*)  take((size_t)NN * HD * 2);   // 16 MB
    float* outF  = (float*)take((size_t)NN * DIM * 4);  //  4 MB (residual 1, f32)
    f16*   outH  = (f16*)  take((size_t)NN * DIM * 2);  //  2 MB
    f16*   h1h   = (f16*)  take((size_t)NN * DIM * 2);
    f16*   h2h   = (f16*)  take((size_t)NN * DIM * 2);
    f16*   featsh= (f16*)  take((size_t)NN * DIM * 2);
    f16*   disth = (f16*)  take((size_t)NN * DIM * 2);
    f16*   WqT   = (f16*)  take((size_t)HD * DIM * 2);  // [1024 x 128]
    f16*   WkT   = (f16*)  take((size_t)HD * DIM * 2);
    f16*   WvT   = (f16*)  take((size_t)HD * DIM * 2);
    f16*   WoT   = (f16*)  take((size_t)DIM * HD * 2);  // [128 x 1024]
    f16*   W1T   = (f16*)  take((size_t)DIM * DIM * 2);
    f16*   W2T   = (f16*)  take((size_t)DIM * DIM * 2);
    f16*   W3T   = (f16*)  take((size_t)DIM * DIM * 2);
    f16*   W4T   = (f16*)  take((size_t)DIM * DIM * 2);

    const int T = 256;
    // ---- precision prep: f16 activations, f16 transposed weights ----
    cvt_f32_to_f16<<<(NN * DIM + T - 1) / T, T, 0, stream>>>(featsh, features, NN * DIM);
    cvt_f32_to_f16<<<(NN * DIM + T - 1) / T, T, 0, stream>>>(disth, distf, NN * DIM);
    transpose_cvt<<<(DIM * HD + T - 1) / T, T, 0, stream>>>(WqT, Wq, DIM, HD);
    transpose_cvt<<<(DIM * HD + T - 1) / T, T, 0, stream>>>(WkT, Wk, DIM, HD);
    transpose_cvt<<<(DIM * HD + T - 1) / T, T, 0, stream>>>(WvT, Wv, DIM, HD);
    transpose_cvt<<<(HD * DIM + T - 1) / T, T, 0, stream>>>(WoT, Wo, HD, DIM);
    transpose_cvt<<<(DIM * DIM + T - 1) / T, T, 0, stream>>>(W1T, W1, DIM, DIM);
    transpose_cvt<<<(DIM * DIM + T - 1) / T, T, 0, stream>>>(W2T, W2, DIM, DIM);
    transpose_cvt<<<(DIM * DIM + T - 1) / T, T, 0, stream>>>(W3T, W3, DIM, DIM);
    transpose_cvt<<<(DIM * DIM + T - 1) / T, T, 0, stream>>>(W4T, W4, DIM, DIM);

    dim3 blk(256);
    // ---- Q/K/V projections: [8192x128] @ [128x1024], WMMA f16 ----
    gemm_wmma<DIM, 0><<<dim3(NN / 128, HD / 64), blk, 0, stream>>>(disth,  WqT, bq, nullptr, nullptr, qh,  NN, HD);
    gemm_wmma<DIM, 0><<<dim3(NN / 128, HD / 64), blk, 0, stream>>>(featsh, WkT, bk, nullptr, nullptr, khb, NN, HD);
    gemm_wmma<DIM, 0><<<dim3(NN / 128, HD / 64), blk, 0, stream>>>(featsh, WvT, bv, nullptr, nullptr, vhb, NN, HD);

    // ---- neighbour attention (gather + softmax), one wave per (node, head) ----
    attn_kernel<<<(NN * HH) / 8, blk, 0, stream>>>(qh, khb, vhb, structure, atth);

    // ---- output projection + residual 1: out = dist + att @ Wo + bo (K=1024, 8 K-chunks, double-buffered) ----
    gemm_wmma<HD, 2><<<dim3(NN / 128, DIM / 64), blk, 0, stream>>>(atth, WoT, bo, distf, outF, outH, NN, DIM);

    // ---- MLP: 3 relu layers + linear out, + residual 2 into d_out ----
    gemm_wmma<DIM, 1><<<dim3(NN / 128, DIM / 64), blk, 0, stream>>>(outH, W1T, b1, nullptr, nullptr, h1h, NN, DIM);
    gemm_wmma<DIM, 1><<<dim3(NN / 128, DIM / 64), blk, 0, stream>>>(h1h,  W2T, b2, nullptr, nullptr, h2h, NN, DIM);
    gemm_wmma<DIM, 1><<<dim3(NN / 128, DIM / 64), blk, 0, stream>>>(h2h,  W3T, b3, nullptr, nullptr, h1h, NN, DIM);
    gemm_wmma<DIM, 3><<<dim3(NN / 128, DIM / 64), blk, 0, stream>>>(h1h,  W4T, b4, outF, (float*)d_out, nullptr, NN, DIM);

    (void)in_sizes; (void)n_in; (void)out_size; (void)ws_size;
}
